// LocalEqFeatureExtraction_61065845014851
// MI455X (gfx1250) — matrix operand order repaired and checked by reading, not compiled, not run on hardware
//
#include <hip/hip_runtime.h>

// LocalEqFeatureExtraction for MI455X (gfx1250, wave32, WMMA).
// One wave per voxel: f32 VALU feature build -> bf16 WMMA 16x16x32 GEMM
// (20x27 @ 27x64, K padded to 32, M padded to 2x16 tiles, N = 4x16 tiles)
// -> fused BN + ReLU + row-max in the D-fragment layout.
//
// v2: element-loop outermost so q-coordinate bpermutes are shared across the
// two A row-tiles (halves ds ops), raw v_sqrt_f32 (no special-case fixup),
// native __bf16 conversions instead of manual RNE packing.

#define V_TOTAL 80000
#define P_PTS   20
#define C_OUTC  64
#define C_IN    27     // 3 coords + 1 pt_feat + 3 cluster + 20 edge
#define BN_EPS  1e-3f

typedef __attribute__((ext_vector_type(16))) __bf16 v16bf;
typedef __attribute__((ext_vector_type(8)))  float  v8f;

__device__ __forceinline__ float wsum32(float x) {
  x += __shfl_xor(x, 16, 32);
  x += __shfl_xor(x,  8, 32);
  x += __shfl_xor(x,  4, 32);
  x += __shfl_xor(x,  2, 32);
  x += __shfl_xor(x,  1, 32);
  return x;
}

__global__ __launch_bounds__(256) void pillar_wmma_kernel(
    const float* __restrict__ features,    // [V][P][4]
    const int*   __restrict__ num_points,  // [V]
    const float* __restrict__ W,           // [27][64]
    const float* __restrict__ gammap,
    const float* __restrict__ betap,
    const float* __restrict__ rmean,
    const float* __restrict__ rvar,
    float* __restrict__ out)               // [V][64]
{
  const int lane   = threadIdx.x & 31;
  const int laneHi = lane >> 4;            // 0: lanes 0-15, 1: lanes 16-31
  const int nlow   = lane & 15;
  const int kbaseA = laneHi * 8;           // A-fragment K base (ISA 16-bit A layout)

  // ---- loop-invariant: bf16 B fragments of W (K=27 padded to 32) + folded BN ----
  v16bf bfrag[4];
  float scale[4], shift[4];
  #pragma unroll
  for (int nt = 0; nt < 4; ++nt) {
    const int n = nt * 16 + nlow;
    const float s = gammap[n] * __frsqrt_rn(rvar[n] + BN_EPS);
    scale[nt] = s;
    shift[nt] = betap[n] - rmean[n] * s;
    #pragma unroll
    for (int e = 0; e < 16; ++e) {
      const int k = laneHi * 8 + ((e < 8) ? e : (8 + e));   // B layout: same K interleave as A
      const float w = (k < C_IN) ? W[k * C_OUTC + n] : 0.0f;
      bfrag[nt][e] = (__bf16)w;
    }
  }

  const int waveId   = (int)((blockIdx.x * blockDim.x + threadIdx.x) >> 5);
  const int numWaves = (int)((gridDim.x * blockDim.x) >> 5);

  for (int v = waveId; v < V_TOTAL; v += numWaves) {
    // ---- load one point per lane (lanes 0..19 valid) ----
    float cx = 0.f, cy = 0.f, cz = 0.f, pf = 0.f;
    if (lane < P_PTS) {
      const float4 f4 = *(const float4*)(features + ((long)v * P_PTS + lane) * 4);
      cx = f4.x; cy = f4.y; cz = f4.z; pf = f4.w;
    }
    // prefetch next voxel's points into the cache hierarchy
    const int vn = v + numWaves;
    if (vn < V_TOTAL && lane < P_PTS)
      __builtin_prefetch(features + ((long)vn * P_PTS + lane) * 4, 0, 3);

    const int   npts = num_points[v];
    const float invn = 1.0f / (float)npts;
    const float mx = wsum32(cx) * invn;
    const float my = wsum32(cy) * invn;
    const float mz = wsum32(cz) * invn;

    // ---- per-tile row coordinates (tile0 rows 0-15, tile1 rows 16-31) ----
    float rx[2], ry[2], rz[2], rf[2], maskR[2], rowOk[2];
    #pragma unroll
    for (int t = 0; t < 2; ++t) {
      const int row = t * 16 + nlow;
      rx[t] = __shfl(cx, row & 31, 32);
      ry[t] = __shfl(cy, row & 31, 32);
      rz[t] = __shfl(cz, row & 31, 32);
      rf[t] = __shfl(pf, row & 31, 32);
      rowOk[t] = (row < P_PTS) ? 1.0f : 0.0f;                 // pad rows -> exact zero
      maskR[t] = (row < P_PTS && row < npts) ? 1.0f : 0.0f;   // node row mask
    }

    // ---- build bf16 A fragments; q-shuffles shared across both row tiles ----
    v16bf afrag0, afrag1;
    #pragma unroll
    for (int e = 0; e < 16; ++e) {
      const int k = kbaseA + ((e < 8) ? e : (8 + e));   // ISA 16-bit A K interleave
      const int q  = k - 7;                              // edge column
      const int qs = ((unsigned)q < P_PTS) ? q : 0;
      const float qx = __shfl(cx, qs, 32);
      const float qy = __shfl(cy, qs, 32);
      const float qz = __shfl(cz, qs, 32);
      const float mQ = ((unsigned)q < (unsigned)npts) ? 1.0f : 0.0f;  // column mask
      #pragma unroll
      for (int t = 0; t < 2; ++t) {
        float node = (k == 0) ? rx[t]
                   : (k == 1) ? ry[t]
                   : (k == 2) ? rz[t]
                   : (k == 3) ? rf[t]
                   : (k == 4) ? (rx[t] - mx)
                   : (k == 5) ? (ry[t] - my)
                   :            (rz[t] - mz);
        const float dx = rx[t] - qx, dy = ry[t] - qy, dz = rz[t] - qz;
        const float d2 = dx * dx + dy * dy + dz * dz;
        const float dist = __builtin_amdgcn_sqrtf(d2);    // sqrt(0)==0: no select needed
        const float xval = (k < 7)    ? node * maskR[t]
                         : (k < C_IN) ? dist * (mQ * rowOk[t])
                         : 0.0f;
        if (t == 0) afrag0[e] = (__bf16)xval;
        else        afrag1[e] = (__bf16)xval;
      }
    }

    // ---- 8x WMMA + fused BN/ReLU/row-max ----
    const v8f czero = {0.f, 0.f, 0.f, 0.f, 0.f, 0.f, 0.f, 0.f};
    #pragma unroll
    for (int nt = 0; nt < 4; ++nt) {
      v8f d0 = __builtin_amdgcn_wmma_f32_16x16x32_bf16(
          false, afrag0, false, bfrag[nt], (short)0, czero, false, false);
      v8f d1 = __builtin_amdgcn_wmma_f32_16x16x32_bf16(
          false, afrag1, false, bfrag[nt], (short)0, czero, false, false);

      float mxv = 0.0f;                     // ReLU => true max >= 0, 0 is a safe identity
      #pragma unroll
      for (int r = 0; r < 8; ++r) {         // tile0 rows 0..15: all valid
        float y = fmaxf(d0[r] * scale[nt] + shift[nt], 0.0f);
        mxv = fmaxf(mxv, y);
      }
      #pragma unroll
      for (int r = 0; r < 4; ++r) {         // tile1: rows 16..19 valid only (laneHi==0)
        float y = fmaxf(d1[r] * scale[nt] + shift[nt], 0.0f);
        y = (laneHi == 0) ? y : 0.0f;
        mxv = fmaxf(mxv, y);
      }
      mxv = fmaxf(mxv, __shfl_xor(mxv, 16, 32));  // fold halves (rows 0-7+16-19 vs 8-15)
      if (lane < 16)
        out[(long)v * C_OUTC + nt * 16 + lane] = mxv;
    }
  }
}

extern "C" void kernel_launch(void* const* d_in, const int* in_sizes, int n_in,
                              void* d_out, int out_size, void* d_ws, size_t ws_size,
                              hipStream_t stream) {
  (void)in_sizes; (void)n_in; (void)out_size; (void)d_ws; (void)ws_size;
  const float* features   = (const float*)d_in[0];
  const int*   num_points = (const int*)  d_in[1];
  // d_in[2] = coors: unused by the reference computation
  const float* W      = (const float*)d_in[3];
  const float* gammap = (const float*)d_in[4];
  const float* betap  = (const float*)d_in[5];
  const float* rmean  = (const float*)d_in[6];
  const float* rvar   = (const float*)d_in[7];
  float* out = (float*)d_out;

  dim3 grid(2048), block(256);   // 16384 waves, ~5 voxels each (grid-stride)
  hipLaunchKernelGGL(pillar_wmma_kernel, grid, block, 0, stream,
                     features, num_points, W, gammap, betap, rmean, rvar, out);
}
